// ShiftedWindowAttention_83038897701188
// MI455X (gfx1250) — compile-verified, matrix operand order checked by reading, not run on hardware
//
#include <hip/hip_runtime.h>

// ---------------------------------------------------------------------------
// Shifted-window attention (Swin): 3 kernels, f16 WMMA / f32 accumulation.
//   prep : weights f32->f16, expand rel-pos bias
//   qkv  : shift+gather -> LDS, GEMM 64x576 (K=192), q/k/v coalesced f16
//   attn : per-window: S=qk^T + bias + mask, softmax, PV, fused proj GEMM,
//          LDS-staged coalesced shifted write-back (no HBM round trip).
// B=8 C=192 H=W=192 WS=8 SS=4 NH=6 HD=32 N=64 ; 4608 windows.
// ---------------------------------------------------------------------------

typedef _Float16 half_t;
typedef _Float16 v16h __attribute__((ext_vector_type(16)));
typedef _Float16 v8h  __attribute__((ext_vector_type(8)));
typedef float    v8f  __attribute__((ext_vector_type(8)));

constexpr int BB   = 8;
constexpr int CC   = 192;
constexpr int HH   = 192;
constexpr int WW   = 192;
constexpr int WSZ  = 8;
constexpr int SSZ  = 4;
constexpr int NHD  = 6;
constexpr int HD   = 32;
constexpr int NTOK = 64;               // WSZ*WSZ
constexpr int NWX  = 24;               // HH/WSZ
constexpr int NWIN = NWX * NWX;        // 576 windows per image
constexpr int BWIN = BB * NWIN;        // 4608 windows total
constexpr int OPAD = CC + 1;           // 193: conflict-free f32 LDS row stride
constexpr float SCALE = 0.17677669529663687f;  // 32^-0.5

// v16h A/B fragment = two contiguous 16-byte runs (ISA 16-bit 16x32 layout:
// lane holds K = kb..kb+7 and kb+16..kb+23, kb = (lane>=16)*8).
__device__ inline v16h make_frag(const half_t* p0, const half_t* p1) {
  v8h lo = *(const v8h*)p0;
  v8h hi = *(const v8h*)p1;
  return __builtin_shufflevector(lo, hi, 0, 1, 2, 3, 4, 5, 6, 7,
                                 8, 9, 10, 11, 12, 13, 14, 15);
}

__device__ inline v8f wmma16(v16h a, v16h b, v8f c) {
  return __builtin_amdgcn_wmma_f32_16x16x32_f16(
      false, a, false, b, (short)0, c, false, false);
}

// ---------------------------------------------------------------------------
// Prep: weights f32->f16, relative-position bias expanded to (6,64,64) f32.
// ---------------------------------------------------------------------------
__global__ __launch_bounds__(256) void swin_prep_kernel(
    const float* __restrict__ qkv_w, const float* __restrict__ proj_w,
    const float* __restrict__ table, half_t* __restrict__ qkvw_h,
    half_t* __restrict__ projw_h, float* __restrict__ bias_f) {
  int i = blockIdx.x * blockDim.x + threadIdx.x;
  if (i < 3 * CC * CC) qkvw_h[i] = (half_t)qkv_w[i];
  if (i < CC * CC) projw_h[i] = (half_t)proj_w[i];
  if (i < NHD * NTOK * NTOK) {
    int h = i / (NTOK * NTOK);
    int rem = i % (NTOK * NTOK);
    int ti = rem / NTOK, tj = rem % NTOK;
    int iy = ti >> 3, ix = ti & 7, jy = tj >> 3, jx = tj & 7;
    int idx = (iy - jy + WSZ - 1) * (2 * WSZ - 1) + (ix - jx + WSZ - 1);
    bias_f[i] = table[idx * NHD + h];
  }
}

// ---------------------------------------------------------------------------
// Stage 1: cyclic shift + window gather + QKV GEMM (M=64, K=192, N=576).
// One block per window, 8 waves x 18 of 144 tiles. q scaled by 32^-0.5.
// q,k,v all stored [win][h][n][d] f16 (32B-segment coalesced stores).
// ---------------------------------------------------------------------------
__global__ __launch_bounds__(256) void swin_qkv_kernel(
    const float* __restrict__ x, const half_t* __restrict__ qkvw_h,
    const float* __restrict__ qkv_b, half_t* __restrict__ q_h,
    half_t* __restrict__ k_h, half_t* __restrict__ v_h) {
  __shared__ __attribute__((aligned(16))) half_t a_lds[NTOK * CC];  // 24 KB

  int win = blockIdx.x;
  int b = win / NWIN;
  int wrem = win % NWIN;
  int wy = wrem / NWX, wx = wrem % NWX;
  int tid = threadIdx.x;

  for (int e = tid; e < NTOK * CC; e += 256) {
    int n = e & 63;
    int c = e >> 6;
    int ty = n >> 3, tx = n & 7;
    int hh = (wy * WSZ + ty + SSZ) % HH;
    int ww = (wx * WSZ + tx + SSZ) % WW;
    a_lds[n * CC + c] = (half_t)x[(((size_t)b * CC + c) * HH + hh) * WW + ww];
  }
  __syncthreads();

  int wave = tid >> 5, lane = tid & 31;
  int l15 = lane & 15;
  int kb2 = (lane >> 4) * 8;
  int half8 = (lane >> 4) << 3;

  for (int t = wave; t < 144; t += 8) {       // 4 mtiles x 36 ntiles
    int mt = t & 3, nt = t >> 2;
    int colc = nt * 16 + l15;
    int m = mt * 16 + l15;
    const half_t* brow = qkvw_h + (size_t)colc * CC;  // B[k][n] = W[n][k]
    v8f acc = {};
#pragma unroll
    for (int ks = 0; ks < 6; ++ks) {
      int k0 = ks * 32;
      v16h a = make_frag(&a_lds[m * CC + k0 + kb2],
                         &a_lds[m * CC + k0 + kb2 + 16]);
      v16h bf = make_frag(brow + k0 + kb2, brow + k0 + kb2 + 16);
      acc = wmma16(a, bf, acc);
    }
    float bias = qkv_b[colc];
    int seg = nt / 12;                        // 0=q 1=k 2=v (tile-uniform)
    int cseg = colc - seg * CC;               // 0..191
    int hh2 = cseg / HD, dd = cseg % HD;
    half_t* dst = (seg == 0) ? q_h : ((seg == 1) ? k_h : v_h);
    float scl = (seg == 0) ? SCALE : 1.0f;
    size_t base = (((size_t)win * NHD + hh2) * NTOK) * HD + dd;
#pragma unroll
    for (int r = 0; r < 8; ++r) {
      int mrow = mt * 16 + r + half8;
      dst[base + (size_t)mrow * HD] = (half_t)((acc[r] + bias) * scl);
    }
  }
}

// ---------------------------------------------------------------------------
// Stage 2 (fused): attention + projection + shifted write-back.
// 6 waves, one head each. LDS: vT 24KB | P 12KB | O(f16) 24KB = 60KB;
// the proj write-back f32 chunk (16 x 193) reuses the vT/P region.
// ---------------------------------------------------------------------------
__global__ __launch_bounds__(192) void swin_attn_kernel(
    const half_t* __restrict__ q_h, const half_t* __restrict__ k_h,
    const half_t* __restrict__ v_h, const float* __restrict__ bias_f,
    const half_t* __restrict__ projw_h, const float* __restrict__ proj_b,
    float* __restrict__ out) {
  __shared__ __attribute__((aligned(16))) char smem[60 * 1024];
  half_t* vt_lds = (half_t*)smem;                          // NHD*HD*NTOK (24KB)
  half_t* p_lds  = (half_t*)(smem + 24 * 1024);            // NHD*16*NTOK (12KB)
  half_t* o16    = (half_t*)(smem + 36 * 1024);            // NTOK*CC     (24KB)
  float*  ochunk = (float*)smem;                           // 16*OPAD, reuse

  int win = blockIdx.x;
  int b = win / NWIN;
  int wrem = win % NWIN;
  int wy = wrem / NWX, wx = wrem % NWX;
  bool edgeR = (wy == NWX - 1), edgeC = (wx == NWX - 1);

  int tid = threadIdx.x;
  int wave = tid >> 5, lane = tid & 31;
  int h = wave;
  int l15 = lane & 15;
  int kb2 = (lane >> 4) * 8;
  int half8 = (lane >> 4) << 3;

  const half_t* qb = q_h + ((size_t)win * NHD + h) * NTOK * HD;
  const half_t* kp = k_h + ((size_t)win * NHD + h) * NTOK * HD;
  const half_t* vb = v_h + ((size_t)win * NHD + h) * NTOK * HD;
  const float* bh = bias_f + h * NTOK * NTOK;
  half_t* vt = vt_lds + h * HD * NTOK;
  half_t* pp = p_lds + h * 16 * NTOK;

  // Transpose this head's V into LDS: coalesced b128 reads, ds scatter.
#pragma unroll
  for (int it = 0; it < 8; ++it) {
    int eh = (it * 32 + lane) * 8;            // element index into [n][d]
    v8h vv = *(const v8h*)(vb + eh);
    int n = eh >> 5;
    int d0 = eh & 31;
#pragma unroll
    for (int j = 0; j < 8; ++j) vt[(d0 + j) * NTOK + n] = vv[j];
  }

  for (int mt = 0; mt < 4; ++mt) {
    int m = mt * 16 + l15;
    v16h aq = make_frag(qb + m * HD + kb2, qb + m * HD + kb2 + 16);
    v8f s[4];
#pragma unroll
    for (int nt = 0; nt < 4; ++nt) {
      v16h bk = make_frag(kp + (nt * 16 + l15) * HD + kb2,
                          kp + (nt * 16 + l15) * HD + kb2 + 16);
      v8f z = {};
      s[nt] = wmma16(aq, bk, z);
    }
    // bias + shift mask (region label: edge ? (coord<4 ? 1 : 2) : 0)
#pragma unroll
    for (int nt = 0; nt < 4; ++nt) {
      int j = nt * 16 + l15;
      int jy = j >> 3, jx = j & 7;
      int labj = (edgeR ? (jy < SSZ ? 1 : 2) : 0) * 3 +
                 (edgeC ? (jx < SSZ ? 1 : 2) : 0);
#pragma unroll
      for (int r = 0; r < 8; ++r) {
        int i = mt * 16 + r + half8;
        int iy = i >> 3, ix = i & 7;
        int labi = (edgeR ? (iy < SSZ ? 1 : 2) : 0) * 3 +
                   (edgeC ? (ix < SSZ ? 1 : 2) : 0);
        s[nt][r] += bh[i * NTOK + j] + ((labi != labj) ? -100.0f : 0.0f);
      }
    }
    // Row softmax: a row lives in one 16-lane half across the 4 nt regs.
#pragma unroll
    for (int r = 0; r < 8; ++r) {
      float mx = fmaxf(fmaxf(s[0][r], s[1][r]), fmaxf(s[2][r], s[3][r]));
      for (int off = 1; off < 16; off <<= 1)
        mx = fmaxf(mx, __shfl_xor(mx, off));
      float sum = 0.f;
#pragma unroll
      for (int nt = 0; nt < 4; ++nt) {
        float e = __expf(s[nt][r] - mx);
        s[nt][r] = e;
        sum += e;
      }
      for (int off = 1; off < 16; off <<= 1) sum += __shfl_xor(sum, off);
      float inv = 1.0f / sum;
#pragma unroll
      for (int nt = 0; nt < 4; ++nt) s[nt][r] *= inv;
    }
    // P rows (16 x 64) for this mtile -> LDS, f16, A-fragment-readable.
#pragma unroll
    for (int nt = 0; nt < 4; ++nt) {
      int j = nt * 16 + l15;
#pragma unroll
      for (int r = 0; r < 8; ++r) pp[(r + half8) * NTOK + j] = (half_t)s[nt][r];
    }
    // O tile = P @ V for this mtile, into head-concatenated LDS buffer.
#pragma unroll
    for (int dt = 0; dt < 2; ++dt) {
      int d = dt * 16 + l15;
      v8f o = {};
#pragma unroll
      for (int ks = 0; ks < 2; ++ks) {
        int k0 = ks * 32;
        v16h ap = make_frag(pp + l15 * NTOK + k0 + kb2,
                            pp + l15 * NTOK + k0 + kb2 + 16);
        v16h bv = make_frag(vt + d * NTOK + k0 + kb2,
                            vt + d * NTOK + k0 + kb2 + 16);
        o = wmma16(ap, bv, o);
      }
#pragma unroll
      for (int r = 0; r < 8; ++r) {
        int mrow = mt * 16 + r + half8;
        o16[mrow * CC + h * HD + d] = (half_t)o[r];
      }
    }
  }

  __syncthreads();  // o16 complete across all heads; vT/P dead from here

  // Fused projection: 64x192, K=192. Per mtile: 12 ntiles over 6 waves,
  // stage f32 in LDS (stride 193 -> conflict-free), coalesced shifted store.
  for (int mt = 0; mt < 4; ++mt) {
    for (int tt = wave; tt < 12; tt += 6) {
      int colc = tt * 16 + l15;
      int m = mt * 16 + l15;
      const half_t* arow = o16 + m * CC;
      const half_t* brow = projw_h + (size_t)colc * CC;
      v8f acc = {};
#pragma unroll
      for (int ks = 0; ks < 6; ++ks) {
        int k0 = ks * 32;
        v16h a = make_frag(arow + k0 + kb2, arow + k0 + kb2 + 16);
        v16h bf = make_frag(brow + k0 + kb2, brow + k0 + kb2 + 16);
        acc = wmma16(a, bf, acc);
      }
      float pb = proj_b[colc];
#pragma unroll
      for (int r = 0; r < 8; ++r)
        ochunk[(r + half8) * OPAD + colc] = acc[r] + pb;
    }
    __syncthreads();
    // Write back 16 token rows: consecutive lanes -> consecutive tx ->
    // 32B-contiguous w segments (with the inverse cyclic shift).
    for (int e = tid; e < 16 * CC; e += 192) {
      int nl = e & 15;
      int c = e >> 4;
      int n = mt * 16 + nl;
      int ty = n >> 3, tx = n & 7;
      int hh = (wy * WSZ + ty + SSZ) % HH;
      int ww = (wx * WSZ + tx + SSZ) % WW;
      out[(((size_t)b * CC + c) * HH + hh) * WW + ww] = ochunk[nl * OPAD + c];
    }
    __syncthreads();
  }
}

// ---------------------------------------------------------------------------
// Launch
// ---------------------------------------------------------------------------
extern "C" void kernel_launch(void* const* d_in, const int* in_sizes, int n_in,
                              void* d_out, int out_size, void* d_ws,
                              size_t ws_size, hipStream_t stream) {
  (void)in_sizes; (void)n_in; (void)out_size; (void)ws_size;
  const float* x      = (const float*)d_in[0];
  const float* qkv_w  = (const float*)d_in[1];
  const float* qkv_b  = (const float*)d_in[2];
  const float* proj_w = (const float*)d_in[3];
  const float* proj_b = (const float*)d_in[4];
  const float* table  = (const float*)d_in[5];
  float* out = (float*)d_out;

  char* ws = (char*)d_ws;
  const size_t SZ_QKVW  = (size_t)3 * CC * CC * sizeof(half_t);      // 221184
  const size_t SZ_PROJW = (size_t)CC * CC * sizeof(half_t);          // 73728
  const size_t SZ_BIAS  = (size_t)NHD * NTOK * NTOK * sizeof(float); // 98304
  const size_t SZ_ONE   = (size_t)BWIN * NHD * NTOK * HD * sizeof(half_t);

  half_t* qkvw_h  = (half_t*)(ws);
  half_t* projw_h = (half_t*)(ws + SZ_QKVW);
  float*  bias_f  = (float*)(ws + SZ_QKVW + SZ_PROJW);
  half_t* q_h     = (half_t*)(ws + SZ_QKVW + SZ_PROJW + SZ_BIAS);
  half_t* k_h     = (half_t*)((char*)q_h + SZ_ONE);
  half_t* v_h     = (half_t*)((char*)k_h + SZ_ONE);

  int prep_n = 3 * CC * CC;
  swin_prep_kernel<<<dim3((prep_n + 255) / 256), dim3(256), 0, stream>>>(
      qkv_w, proj_w, table, qkvw_h, projw_h, bias_f);
  swin_qkv_kernel<<<dim3(BWIN), dim3(256), 0, stream>>>(
      x, qkvw_h, qkv_b, q_h, k_h, v_h);
  swin_attn_kernel<<<dim3(BWIN), dim3(192), 0, stream>>>(
      q_h, k_h, v_h, bias_f, projw_h, proj_b, out);
}
